// YOLOXHead_27814208209543
// MI455X (gfx1250) — compile-verified, matrix operand order checked by reading, not run on hardware
//
#include <hip/hip_runtime.h>
#include <hip/hip_bf16.h>
#include <cstdint>
#include <cstddef>

typedef __attribute__((ext_vector_type(16))) __bf16 v16bf;
typedef __attribute__((ext_vector_type(8)))  float  v8f;

#define THREADS 256

#if defined(__AMDGCN__) && __has_builtin(__builtin_amdgcn_tensor_load_to_lds)
#define USE_TDM 1
#else
#define USE_TDM 0
#endif

// ---------- helpers ----------
__device__ __forceinline__ unsigned short f2bf(float f) {
  unsigned int u = __float_as_uint(f);
  u += 0x7FFFu + ((u >> 16) & 1u);   // round-to-nearest-even to bf16
  return (unsigned short)(u >> 16);
}

__device__ __forceinline__ float dev_sigmoid(float x) {
  return 1.0f / (1.0f + __expf(-x));
}

// ---------- elementwise / packing kernels ----------
// NCHW f32 -> NHWC bf16
__global__ __launch_bounds__(THREADS) void cvt_nhwc_kernel(const float* __restrict__ in,
                                                           unsigned short* __restrict__ out,
                                                           int C, int HW, int total) {
  int idx = blockIdx.x * blockDim.x + threadIdx.x;   // NCHW linear (coalesced read)
  if (idx >= total) return;
  int hw = idx % HW;
  int t  = idx / HW;
  int c  = t % C;
  int b  = t / C;
  out[((size_t)b * HW + hw) * C + c] = f2bf(in[idx]);
}

// OIHW weight -> bf16 BT[Npad rows][K] (n-major), k = r*Cin + ci (r = kh*KS+kw),
// zero-padded rows for n >= Cout.
__global__ __launch_bounds__(THREADS) void pack_weight_kernel(const float* __restrict__ w,
                                                              unsigned short* __restrict__ BT,
                                                              int K, int Cout, int Npad,
                                                              int KSS, int cinShift) {
  int idx = blockIdx.x * blockDim.x + threadIdx.x;   // n-major: idx = n*K + k
  if (idx >= Npad * K) return;
  int n = idx / K;
  int k = idx - n * K;
  unsigned short v = 0;
  if (n < Cout) {
    int r  = k >> cinShift;
    int ci = k & ((1 << cinShift) - 1);
    v = f2bf(w[(size_t)n * K + ci * KSS + r]);
  }
  BT[idx] = v;
}

// Pred B matrices (n-major, K=256): BroT rows = [reg(4) | obj(1) | 0...], BclsT = [cls(80) | 0...]
__global__ __launch_bounds__(THREADS) void pack_pred_kernel(const float* __restrict__ reg_w,
                                                            const float* __restrict__ obj_w,
                                                            const float* __restrict__ cls_w,
                                                            unsigned short* __restrict__ BroT,
                                                            unsigned short* __restrict__ BclsT) {
  int idx = blockIdx.x * blockDim.x + threadIdx.x;   // 128 * 256
  if (idx >= 128 * 256) return;
  int n = idx >> 8;
  int k = idx & 255;
  unsigned short v1 = 0, v2 = 0;
  if (n < 4)       v1 = f2bf(reg_w[n * 256 + k]);
  else if (n == 4) v1 = f2bf(obj_w[k]);
  if (n < 80)      v2 = f2bf(cls_w[n * 256 + k]);
  BroT[idx]  = v1;
  BclsT[idx] = v2;
}

__global__ void pack_bias_kernel(const float* __restrict__ reg_b,
                                 const float* __restrict__ obj_b,
                                 float* __restrict__ out5) {
  int i = threadIdx.x;
  if (i < 4)       out5[i] = reg_b[i];
  else if (i == 4) out5[4] = obj_b[0];
}

// ---------- fragment loader (CDNA5 16-bit 16x32 operand layout) ----------
// Tile row-major in LDS, K pairs contiguous -> 8 aligned b32 loads per fragment.
__device__ __forceinline__ v16bf load_frag(const unsigned short* __restrict__ base,
                                           int ldStride, int lane) {
  union { v16bf v; unsigned int u[8]; } f;
  const unsigned short* p = base + (lane & 15) * ldStride;
  int hi = (lane >> 4) << 3;
#pragma unroll
  for (int r = 0; r < 8; ++r) {
    int k0 = ((r >> 2) << 4) + hi + ((r & 3) << 1);
    f.u[r] = *(const unsigned int*)(p + k0);   // even stride + even k0 -> 4B aligned
  }
  return f.v;
}

// ---------- TDM issue for the B-weight tile ----------
// BT is n-major [N][K]; tile = 64 contiguous k x 128 n-rows (row stride K elements).
// LDS padding: 1 dword after every 32 dwords (=1 row of 128B) -> row stride 66 ushorts.
#if USE_TDM
typedef __attribute__((ext_vector_type(4))) unsigned int tdm_v4u;
typedef __attribute__((ext_vector_type(8))) int tdm_v8i;
typedef __attribute__((ext_vector_type(4))) int tdm_v4i;

__device__ __forceinline__ void tdm_load_b_tile(const unsigned short* gptr,
                                                unsigned int ldsOff, int K) {
  unsigned long long ga = (unsigned long long)(size_t)gptr;
  tdm_v4u g0;
  g0[0] = 1u;                                            // count=1, user mode, gather off
  g0[1] = ldsOff;                                        // lds_addr (bytes)
  g0[2] = (unsigned int)ga;                              // global_addr[31:0]
  g0[3] = (unsigned int)((ga >> 32) & 0x1FFFFFFull)      // global_addr[56:32]
          | (2u << 30);                                  // type = 2 ("image")
  unsigned int td0 = (unsigned int)K;                    // tensor_dim0 (elements)
  unsigned int td1 = 0x7FFFFFFFu;                        // tensor_dim1: never OOB
  tdm_v8i g1;
  g1[0] = (int)((1u << 16)                               // data_size = 1 (2 bytes)
          | (1u << 20)                                   // pad_enable
          | (4u << 22));                                 // pad_interval: 32 dwords
                                                         // pad_amount: 0 -> 1 dword
  g1[1] = (int)((td0 & 0xFFFFu) << 16);                  // tensor_dim0[15:0] @ bits[63:48]
  g1[2] = (int)((td0 >> 16) | ((td1 & 0xFFFFu) << 16));  // dim0 hi | dim1 lo
  g1[3] = (int)((td1 >> 16) | (64u << 16));              // dim1 hi | tile_dim0 = 64
  g1[4] = (int)128u;                                     // tile_dim1 = 128, tile_dim2 = 0
  g1[5] = (int)td0;                                      // tensor_dim0_stride[31:0] = K
  g1[6] = 0;
  g1[7] = 0;
  tdm_v4i gz = {0, 0, 0, 0};
#if __clang_major__ >= 23
  tdm_v8i gz8 = {0, 0, 0, 0, 0, 0, 0, 0};
  __builtin_amdgcn_tensor_load_to_lds(g0, g1, gz, gz, gz8, 0);
#else
  __builtin_amdgcn_tensor_load_to_lds(g0, g1, gz, gz, 0);
#endif
}
#endif

// ---------- implicit-GEMM conv, NHWC, KS = 1 or 3 (SAME), double-buffered ----------
// A: bf16 NHWC activation; BT: bf16 [Npad][K] n-major, K = KS*KS*Cin, k = r*Cin+ci.
// mode 0: y = y*gamma+beta; y *= sigmoid(y); store bf16 NHWC.
// mode 1: y += beta; store f32 at [m][Cbase+c] of an [M][Ctot] buffer.
template <int KS>
__global__ __launch_bounds__(THREADS) void conv_wmma_kernel(
    const unsigned short* __restrict__ Abf,
    const unsigned short* __restrict__ BT,
    const float* __restrict__ gamma,
    const float* __restrict__ beta,
    unsigned short* __restrict__ out_bf,
    float* __restrict__ out_f32,
    int Batch, int cinShift, int H, int W,
    int Cout, int Ctot, int Cbase, int mode)
{
  __shared__ __align__(16) unsigned short As[2][128][72];  // 128 m-rows x 64 k (+pad)
  __shared__ __align__(16) unsigned short Bs[2][128][66];  // 128 n-rows x 64 k (+TDM pad)

  const int HW = H * W;
  const int M  = Batch * HW;
  const int K  = (KS * KS) << cinShift;
  const int cinMask = (1 << cinShift) - 1;
  const int nslab = K >> 6;

  const int m0 = blockIdx.x * 128;
  const int n0 = blockIdx.y * 128;

  const int tid  = threadIdx.x;
  const int lane = tid & 31;
  const int wave = tid >> 5;
  const int wm0  = (wave & 3) * 32;
  const int wn0  = (wave >> 2) * 64;

  v8f acc[2][4] = {};

  // A staging: thread fills 32 contiguous k of one m-row
  const int arow = tid >> 1;
  const int ah   = (tid & 1) << 5;       // 0 or 32
  const int m    = m0 + arow;
  const int ab   = m / HW;
  const int ahw  = m - ab * HW;
  const int ax_h = ahw / W;
  const int ax_w = ahw - ax_h * W;
  const bool mok = (m < M);

  auto stageA = [&](int buf, int kb) {
    const int ci0 = (kb + ah) & cinMask;         // multiple of 32
    int ih = ax_h, iw = ax_w;
    if (KS == 3) {
      int r  = kb >> cinShift;                   // tap index, uniform over slab
      int kh = r / 3;
      int kw = r - kh * 3;
      ih += kh - 1;
      iw += kw - 1;
    }
    const bool ok = mok && (KS == 1 || (ih >= 0 && ih < H && iw >= 0 && iw < W));
    uint4* as4 = (uint4*)&As[buf][arow][ah];
    if (ok) {
      const size_t srow = (size_t)(ab * H + ih) * W + iw;
      const uint4* ap4 = (const uint4*)(Abf + (srow << cinShift) + ci0);
#pragma unroll
      for (int q = 0; q < 4; ++q) as4[q] = ap4[q];
    } else {
      uint4 z = {0u, 0u, 0u, 0u};
#pragma unroll
      for (int q = 0; q < 4; ++q) as4[q] = z;
    }
  };

  auto stageB = [&](int buf, int kb) {
#if USE_TDM
    if (wave == 0)
      tdm_load_b_tile(BT + (size_t)n0 * K + kb,
                      (unsigned int)(size_t)&Bs[buf][0][0], K);
#else
    const int row  = tid >> 1;
    const int kseg = (tid & 1) << 5;
    const unsigned int* gp32 = (const unsigned int*)(BT + (size_t)(n0 + row) * K + kb + kseg);
    unsigned int* dp32 = (unsigned int*)&Bs[buf][row][kseg];
#pragma unroll
    for (int q = 0; q < 16; ++q) dp32[q] = gp32[q];
#endif
  };

  auto tdmWait = [&]() {
#if USE_TDM
#if __has_builtin(__builtin_amdgcn_s_wait_tensorcnt)
    if (wave == 0) __builtin_amdgcn_s_wait_tensorcnt(0);
#endif
#endif
  };

  auto computeSlab = [&](int buf) {
#pragma unroll
    for (int kk = 0; kk < 64; kk += 32) {
      v16bf a0 = load_frag(&As[buf][wm0][kk], 72, lane);
      v16bf a1 = load_frag(&As[buf][wm0 + 16][kk], 72, lane);
#pragma unroll
      for (int ni = 0; ni < 4; ++ni) {
        v16bf bf = load_frag(&Bs[buf][wn0 + ni * 16][kk], 66, lane);
#if defined(__AMDGCN__)
        acc[0][ni] = __builtin_amdgcn_wmma_f32_16x16x32_bf16(
            false, a0, false, bf, (short)0, acc[0][ni], false, false);
        acc[1][ni] = __builtin_amdgcn_wmma_f32_16x16x32_bf16(
            false, a1, false, bf, (short)0, acc[1][ni], false, false);
#endif
      }
    }
  };

  // ---- prologue: fill buffer 0 ----
  stageB(0, 0);
  stageA(0, 0);
  tdmWait();
  __syncthreads();

  // ---- pipelined main loop: stage slab s+1 while computing slab s ----
  for (int s = 0; s < nslab; ++s) {
    const int cur  = s & 1;
    const int nxt  = cur ^ 1;
    const bool more = (s + 1) < nslab;
    if (more) {
      stageB(nxt, (s + 1) << 6);
      stageA(nxt, (s + 1) << 6);
    }
    computeSlab(cur);
    if (more) tdmWait();
    __syncthreads();
  }

  // ---- epilogue (C/D layout: VGPR r -> M = r + 8*(lane>=16), N = lane&15) ----
  const int lr = lane & 15;
  const int lh = (lane >> 4) << 3;
#pragma unroll
  for (int mi = 0; mi < 2; ++mi) {
#pragma unroll
    for (int ni = 0; ni < 4; ++ni) {
      int tm = m0 + wm0 + mi * 16;
      int tn = n0 + wn0 + ni * 16 + lr;
#pragma unroll
      for (int r = 0; r < 8; ++r) {
        int mm = tm + lh + r;
        if (mm < M && tn < Cout) {
          float y = acc[mi][ni][r];
          if (mode == 0) {
            y = y * gamma[tn] + beta[tn];
            y = y * dev_sigmoid(y);                          // SiLU
            out_bf[(size_t)mm * Cout + tn] = f2bf(y);        // NHWC
          } else {
            y += beta[tn];
            out_f32[(size_t)mm * Ctot + (Cbase + tn)] = y;   // [M][Ctot]
          }
        }
      }
    }
  }
}

// ---------- final decode: grids/strides + sigmoid/exp + concat ----------
// pred buffers are [B][HW][85] f32 per level.
__global__ __launch_bounds__(THREADS) void decode_kernel(
    const float* __restrict__ p0, const float* __restrict__ p1, const float* __restrict__ p2,
    float* __restrict__ out)
{
  int idx = blockIdx.x * blockDim.x + threadIdx.x;
  const int TOT = 8 * 8400 * 85;
  if (idx >= TOT) return;
  int c = idx % 85;
  int t = idx / 85;
  int a = t % 8400;
  int b = t / 8400;

  const float* p; int hw, Wl, st, HWl;
  if (a < 6400)      { p = p0; hw = a;        Wl = 80; st = 8;  HWl = 6400; }
  else if (a < 8000) { p = p1; hw = a - 6400; Wl = 40; st = 16; HWl = 1600; }
  else               { p = p2; hw = a - 8000; Wl = 20; st = 32; HWl = 400;  }

  float v = p[((size_t)b * HWl + hw) * 85 + c];
  float o;
  if (c == 0)      o = (v + (float)(hw % Wl)) * (float)st;
  else if (c == 1) o = (v + (float)(hw / Wl)) * (float)st;
  else if (c < 4)  o = __expf(v) * (float)st;
  else             o = dev_sigmoid(v);
  out[idx] = o;
}

// ---------- host launcher ----------
extern "C" void kernel_launch(void* const* d_in, const int* in_sizes, int n_in,
                              void* d_out, int out_size, void* d_ws, size_t ws_size,
                              hipStream_t stream)
{
  (void)in_sizes; (void)n_in; (void)out_size; (void)ws_size;

  const float* x[3]   = {(const float*)d_in[0], (const float*)d_in[1], (const float*)d_in[2]};
  const int CinL[3]   = {256, 512, 1024};
  const int CinSh[3]  = {8, 9, 10};
  const int Hs[3]     = {80, 40, 20};

  struct ConvP { const float *w, *g, *b; };
  ConvP stem[3], cls0[3], cls1[3], reg0[3], reg1[3];
  const float *cls_pw[3], *cls_pb[3], *reg_pw[3], *reg_pb[3], *obj_pw[3], *obj_pb[3];

  int pi = 3;
  for (int l = 0; l < 3; ++l) {
    auto grab = [&](ConvP& cp) {
      cp.w = (const float*)d_in[pi++];
      cp.g = (const float*)d_in[pi++];
      cp.b = (const float*)d_in[pi++];
    };
    grab(stem[l]); grab(cls0[l]); grab(cls1[l]); grab(reg0[l]); grab(reg1[l]);
    cls_pw[l] = (const float*)d_in[pi++]; cls_pb[l] = (const float*)d_in[pi++];
    reg_pw[l] = (const float*)d_in[pi++]; reg_pb[l] = (const float*)d_in[pi++];
    obj_pw[l] = (const float*)d_in[pi++]; obj_pb[l] = (const float*)d_in[pi++];
  }

  size_t off = 0;
  auto alloc = [&](size_t bytes) -> void* {
    void* p = (char*)d_ws + off;
    off = (off + bytes + 255) & ~(size_t)255;
    return p;
  };

  const float* predBuf[3];
  for (int l = 0; l < 3; ++l) {
    const int Cin = CinL[l], csh = CinSh[l];
    const int H = Hs[l], W = Hs[l], HW = H * W;
    const int M = 8 * HW;
    const int Kst = Cin;
    const int K33 = 256 * 9;

    unsigned short* xbf  = (unsigned short*)alloc((size_t)8 * Cin * HW * 2);
    unsigned short* sbf  = (unsigned short*)alloc((size_t)8 * 256 * HW * 2);
    unsigned short* tbf  = (unsigned short*)alloc((size_t)8 * 256 * HW * 2);
    unsigned short* cfbf = (unsigned short*)alloc((size_t)8 * 256 * HW * 2);
    unsigned short* rfbf = (unsigned short*)alloc((size_t)8 * 256 * HW * 2);
    float* pred          = (float*)alloc((size_t)8 * 85 * HW * 4);
    unsigned short* Bstem = (unsigned short*)alloc((size_t)256 * Kst * 2);
    unsigned short* Bc0   = (unsigned short*)alloc((size_t)256 * K33 * 2);
    unsigned short* Bc1   = (unsigned short*)alloc((size_t)256 * K33 * 2);
    unsigned short* Br0   = (unsigned short*)alloc((size_t)256 * K33 * 2);
    unsigned short* Br1   = (unsigned short*)alloc((size_t)256 * K33 * 2);
    unsigned short* Bro   = (unsigned short*)alloc((size_t)128 * 256 * 2);
    unsigned short* Bcls  = (unsigned short*)alloc((size_t)128 * 256 * 2);
    float* biasro         = (float*)alloc(256);
    predBuf[l] = pred;

    int nxe = 8 * Cin * HW;
    cvt_nhwc_kernel<<<(nxe + THREADS - 1) / THREADS, THREADS, 0, stream>>>(x[l], xbf, Cin, HW, nxe);
    pack_weight_kernel<<<((256 * Kst) + THREADS - 1) / THREADS, THREADS, 0, stream>>>(stem[l].w, Bstem, Kst, 256, 256, 1, csh);
    pack_weight_kernel<<<((256 * K33) + THREADS - 1) / THREADS, THREADS, 0, stream>>>(cls0[l].w, Bc0, K33, 256, 256, 9, 8);
    pack_weight_kernel<<<((256 * K33) + THREADS - 1) / THREADS, THREADS, 0, stream>>>(cls1[l].w, Bc1, K33, 256, 256, 9, 8);
    pack_weight_kernel<<<((256 * K33) + THREADS - 1) / THREADS, THREADS, 0, stream>>>(reg0[l].w, Br0, K33, 256, 256, 9, 8);
    pack_weight_kernel<<<((256 * K33) + THREADS - 1) / THREADS, THREADS, 0, stream>>>(reg1[l].w, Br1, K33, 256, 256, 9, 8);
    pack_pred_kernel<<<((128 * 256) + THREADS - 1) / THREADS, THREADS, 0, stream>>>(reg_pw[l], obj_pw[l], cls_pw[l], Bro, Bcls);
    pack_bias_kernel<<<1, 32, 0, stream>>>(reg_pb[l], obj_pb[l], biasro);

    dim3 blk(THREADS);
    dim3 g256((M + 127) / 128, 2);
    dim3 g128((M + 127) / 128, 1);

    // stem (1x1) + BN + SiLU
    conv_wmma_kernel<1><<<g256, blk, 0, stream>>>(xbf, Bstem, stem[l].g, stem[l].b,
                                                  sbf, nullptr, 8, csh, H, W, 256, 0, 0, 0);
    // cls branch
    conv_wmma_kernel<3><<<g256, blk, 0, stream>>>(sbf, Bc0, cls0[l].g, cls0[l].b,
                                                  tbf, nullptr, 8, 8, H, W, 256, 0, 0, 0);
    conv_wmma_kernel<3><<<g256, blk, 0, stream>>>(tbf, Bc1, cls1[l].g, cls1[l].b,
                                                  cfbf, nullptr, 8, 8, H, W, 256, 0, 0, 0);
    // reg branch
    conv_wmma_kernel<3><<<g256, blk, 0, stream>>>(sbf, Br0, reg0[l].g, reg0[l].b,
                                                  tbf, nullptr, 8, 8, H, W, 256, 0, 0, 0);
    conv_wmma_kernel<3><<<g256, blk, 0, stream>>>(tbf, Br1, reg1[l].g, reg1[l].b,
                                                  rfbf, nullptr, 8, 8, H, W, 256, 0, 0, 0);
    // predictions: reg+obj (5 ch) from reg feature, cls (80 ch) from cls feature
    conv_wmma_kernel<1><<<g128, blk, 0, stream>>>(rfbf, Bro, nullptr, biasro,
                                                  nullptr, pred, 8, 8, H, W, 5, 85, 0, 1);
    conv_wmma_kernel<1><<<g128, blk, 0, stream>>>(cfbf, Bcls, nullptr, cls_pb[l],
                                                  nullptr, pred, 8, 8, H, W, 80, 85, 5, 1);
  }

  int tot = 8 * 8400 * 85;
  decode_kernel<<<(tot + THREADS - 1) / THREADS, THREADS, 0, stream>>>(
      predBuf[0], predBuf[1], predBuf[2], (float*)d_out);
}
